// NeuralMemory_5970004541514
// MI455X (gfx1250) — matrix operand order repaired and checked by reading, not compiled, not run on hardware
//
#include <hip/hip_runtime.h>

// Problem constants (match reference)
#define B_   4
#define N_   8192
#define D_   768
#define H_   12
#define DH_  64
#define C_   64
#define NC_  128             // N / C
#define BN_  (B_ * N_)       // 32768
#define BH_  (B_ * H_)       // 48
#define KVW_ (2 * H_ * DH_)  // 1536
#define MAX_LR_ 0.01f
#define EPS_ 1e-6f

typedef __attribute__((ext_vector_type(16))) __bf16 v16bf;
typedef __attribute__((ext_vector_type(8)))  float  v8f;

__device__ __forceinline__ v8f wmma_bf16f32(v16bf a, v16bf b, v8f c) {
  // D = A(16x32 bf16) x B(32x16 bf16) + C(16x16 f32)
  return __builtin_amdgcn_wmma_f32_16x16x32_bf16(false, a, false, b, (short)0, c,
                                                 false, false);
}

// Fast sigmoid: v_exp_f32 + v_add_f32 + v_rcp_f32 (no IEEE division sequence).
__device__ __forceinline__ float sigf_(float x) {
  return __builtin_amdgcn_rcpf(1.0f + __expf(-x));
}
// Fast rsqrt: single v_rsq_f32.
__device__ __forceinline__ float rsqf_(float x) { return __builtin_amdgcn_rsqf(x); }

// ---- WMMA fragment loaders (contiguous-K order in A and B, wave32) ----
// A 16x32 row-major: lanes 0-15 row M=lane, K 0..15; lanes 16-31 row
// M=lane-16, K 16..31. 16 contiguous K -> 2x ds_load_b128.
__device__ __forceinline__ v16bf ldA(const __bf16* s, int ld, int lane) {
  int row = lane & 15;
  int kb = (lane & 16) ? 16 : 0;
  const __bf16* p = s + row * ld + kb;
  union { v16bf v; uint4 q[2]; } f;
  f.q[0] = *(const uint4*)(p);
  f.q[1] = *(const uint4*)(p + 8);
  return f.v;
}
// B 32x16 from N-major storage Bt[n][k] (Bt[n*ld+k] == B[k][n]):
// lanes 0-15 col N=lane K=0..15; lanes 16-31 col N=lane-16 K=16..31.
// 16 contiguous K -> 2x ds_load_b128.
__device__ __forceinline__ v16bf ldBT(const __bf16* s, int ld, int lane) {
  int col = lane & 15;
  int kb = (lane & 16) ? 16 : 0;
  const __bf16* p = s + col * ld + kb;
  union { v16bf v; uint4 q[2]; } f;
  f.q[0] = *(const uint4*)(p);
  f.q[1] = *(const uint4*)(p + 8);
  return f.v;
}

// 64x64x64 tile matmul: A row-major [64][64], Bt N-major [64][64].
// 8 waves: wm = (w>>1)*16 row tile, wn = (w&1)*32 cols (2 N-tiles each).
__device__ __forceinline__ void mm64(const __bf16* Am, const __bf16* Btn,
                                     int lane, int wm, int wn, v8f acc[2]) {
#pragma unroll
  for (int k0 = 0; k0 < 64; k0 += 32) {
    v16bf af = ldA(Am + wm * 64 + k0, 64, lane);
#pragma unroll
    for (int tn = 0; tn < 2; ++tn) {
      v16bf bf_ = ldBT(Btn + (wn + tn * 16) * 64 + k0, 64, lane);
      acc[tn] = wmma_bf16f32(af, bf_, acc[tn]);
    }
  }
}

// ---------------- Stage kernels ----------------

// u = rmsnorm(seq)*store_g, ur = rmsnorm(seq)*retr_g  -> bf16
__global__ __launch_bounds__(256) void rmsnorm_kernel(
    const float* __restrict__ seq, const float* __restrict__ sg,
    const float* __restrict__ rg, __bf16* __restrict__ ub,
    __bf16* __restrict__ urb) {
  __shared__ float red[256];
  const int row = blockIdx.x;
  const float* x = seq + (size_t)row * D_;
  float ss = 0.f;
  for (int i = threadIdx.x; i < D_; i += 256) { float v = x[i]; ss += v * v; }
  red[threadIdx.x] = ss;
  __syncthreads();
  for (int s2 = 128; s2 > 0; s2 >>= 1) {
    if (threadIdx.x < s2) red[threadIdx.x] += red[threadIdx.x + s2];
    __syncthreads();
  }
  float rs = rsqf_(red[0] * (1.0f / (float)D_) + EPS_);
  for (int i = threadIdx.x; i < D_; i += 256) {
    float v = x[i] * rs;
    ub[(size_t)row * D_ + i]  = (__bf16)(v * sg[i]);
    urb[(size_t)row * D_ + i] = (__bf16)(v * rg[i]);
  }
}

// chunked[b,cn,:] = mean over C tokens of u
__global__ __launch_bounds__(256) void chunk_mean_kernel(
    const __bf16* __restrict__ ub, float* __restrict__ chunked) {
  const int bc = blockIdx.x;  // b*NC + cn
  for (int col = threadIdx.x; col < D_; col += 256) {
    float s = 0.f;
    for (int c = 0; c < C_; ++c)
      s += (float)ub[((size_t)bc * C_ + c) * D_ + col];
    chunked[(size_t)bc * D_ + col] = s * (1.0f / (float)C_);
  }
}

__global__ __launch_bounds__(256) void f2bf_kernel(const float* __restrict__ src,
                                                   __bf16* __restrict__ dst, int n) {
  int i = blockIdx.x * 256 + threadIdx.x;
  if (i < n) dst[i] = (__bf16)src[i];
}

// src [K, Ncols] f32 row-major -> dst [Ncols, K] bf16 (weight pre-transpose)
__global__ __launch_bounds__(256) void f2bf_t_kernel(const float* __restrict__ src,
                                                     __bf16* __restrict__ dst,
                                                     int K, int Ncols) {
  int i = blockIdx.x * 256 + threadIdx.x;
  if (i < K * Ncols) {
    int k = i / Ncols, n = i % Ncols;
    dst[(size_t)n * K + k] = (__bf16)src[i];
  }
}

// out[m,j] = scale * sigmoid(dot(A[m,:], W[:,j])), W is [768,12] row-major
__global__ __launch_bounds__(192) void small_proj_kernel(
    const __bf16* __restrict__ A, const float* __restrict__ W,
    float* __restrict__ out, int M, float scale) {
  int row = blockIdx.x * 16 + threadIdx.x / H_;
  int j = threadIdx.x % H_;
  if (row >= M) return;
  const __bf16* a = A + (size_t)row * D_;
  float s = 0.f;
  for (int d2 = 0; d2 < D_; ++d2) s += (float)a[d2] * W[d2 * H_ + j];
  out[(size_t)row * H_ + j] = scale * sigf_(s);
}

// Generic bf16 WMMA GEMM: C[M,Nc] = A[M,K] @ B[K,Nc], with B supplied
// pre-transposed BwT [Nc,K]. Block tile 128x64, 8 waves, all-b128 LDS traffic.
template <bool OUT_BF16>
__global__ __launch_bounds__(256) void gemm_bf16_kernel(
    const __bf16* __restrict__ A, const __bf16* __restrict__ BwT,
    void* __restrict__ Cout, int M, int K, int Ncols) {
  __shared__ __bf16 sA[128 * 32];
  __shared__ __bf16 sBt[64 * 32];   // N-major
  const int bm = blockIdx.x * 128;
  const int bn = blockIdx.y * 64;
  const int tid = threadIdx.x, lane = tid & 31, wave = tid >> 5;
  const int wm = (wave >> 1) * 32, wn = (wave & 1) * 32;
  const int lhi = (lane & 16) ? 8 : 0, l15 = lane & 15;
  v8f acc[2][2] = {};
  for (int k0 = 0; k0 < K; k0 += 32) {
    __syncthreads();
#pragma unroll
    for (int r = 0; r < 2; ++r) {
      int idx = (tid * 2 + r) * 8;
      int row = idx >> 5, col = idx & 31;
      *(uint4*)&sA[row * 32 + col] =
          *(const uint4*)&A[(size_t)(bm + row) * K + k0 + col];
    }
    {
      int idx = tid * 8;
      int n = idx >> 5, kcol = idx & 31;
      *(uint4*)&sBt[n * 32 + kcol] =
          *(const uint4*)&BwT[(size_t)(bn + n) * K + k0 + kcol];
    }
    if (k0 + 32 < K) {  // prefetch next K tile (global_prefetch_b8)
      __builtin_prefetch(&A[(size_t)(bm + (tid >> 1)) * K + k0 + 32], 0, 1);
      __builtin_prefetch(&BwT[(size_t)(bn + (tid & 63)) * K + k0 + 32], 0, 1);
    }
    __syncthreads();
    v16bf bfr[2];
#pragma unroll
    for (int tn = 0; tn < 2; ++tn)
      bfr[tn] = ldBT(&sBt[(wn + tn * 16) * 32], 32, lane);
#pragma unroll
    for (int tm = 0; tm < 2; ++tm) {
      v16bf af = ldA(&sA[(wm + tm * 16) * 32], 32, lane);
#pragma unroll
      for (int tn = 0; tn < 2; ++tn)
        acc[tm][tn] = wmma_bf16f32(af, bfr[tn], acc[tm][tn]);
    }
  }
#pragma unroll
  for (int tm = 0; tm < 2; ++tm)
#pragma unroll
    for (int tn = 0; tn < 2; ++tn)
#pragma unroll
      for (int r = 0; r < 8; ++r) {
        int row = bm + wm + tm * 16 + r + lhi;
        int col = bn + wn + tn * 16 + l15;
        if (OUT_BF16)
          ((__bf16*)Cout)[(size_t)row * Ncols + col] = (__bf16)acc[tm][tn][r];
        else
          ((float*)Cout)[(size_t)row * Ncols + col] = acc[tm][tn][r];
      }
}

// Fused per-(b,h) memory update: chunk grads (fixed mem_w1/w2, WMMA) + momentum
// and decay recurrences in LDS; emits per-chunk retrieval weights (bf16).
// All matmul operands kept in the orientation their consumer needs.
__global__ __launch_bounds__(256) void memory_update_kernel(
    const __bf16* __restrict__ kvb, const float* __restrict__ lrbuf,
    const float* __restrict__ mombuf, const float* __restrict__ decaybuf,
    const float* __restrict__ mem_w1, const float* __restrict__ mem_w2,
    __bf16* __restrict__ w1c, __bf16* __restrict__ w2c) {
  __shared__ __bf16 s_w1T[4096], s_w2T[4096], s_w2[4096];  // fixed weights
  __shared__ float  s_W1[4096], s_W2[4096];                // evolving weights
  __shared__ float  s_m1[4096], s_m2[4096];                // momentum state
  __shared__ __bf16 s_k[4096], s_kT[4096];
  __shared__ __bf16 s_vdh[4096];   // v (until dpred built), then dh^T
  __shared__ __bf16 s_hb[4096];    // pre-activation (bf16)
  __shared__ __bf16 s_a[4096], s_aT[4096];
  __shared__ __bf16 s_dp[4096], s_dpT[4096];
  __shared__ float  s_lr[64];

  const int bh = blockIdx.x;
  const int b = bh / H_, h = bh % H_;
  const int tid = threadIdx.x, lane = tid & 31, wave = tid >> 5;
  const int wm = (wave >> 1) * 16, wn = (wave & 1) * 32;
  const int lhi = (lane & 16) ? 8 : 0, l15 = lane & 15;

  for (int i = tid; i < 4096; i += 256) {
    int k = i >> 6, n = i & 63;
    float a1 = mem_w1[i], a2 = mem_w2[i];
    s_w1T[n * 64 + k] = (__bf16)a1;   // N-major for ldBT
    s_w2T[n * 64 + k] = (__bf16)a2;   // N-major for ldBT
    s_w2[i] = (__bf16)a2;             // row-major == Bt layout of w2^T
    s_W1[i] = a1; s_W2[i] = a2;
    s_m1[i] = 0.f; s_m2[i] = 0.f;
  }
  __syncthreads();

  for (int j = 0; j < NC_; ++j) {
    // Weights used to retrieve chunk j = state BEFORE this chunk's update.
    size_t wbase = ((size_t)bh * NC_ + j) * 4096;
    for (int i = tid; i < 4096; i += 256) {
      w1c[wbase + i] = (__bf16)s_W1[i];
      w2c[wbase + i] = (__bf16)s_W2[i];
    }
    // Stage k/v chunk [64 tokens x 64 dims]; vectorized b128 global loads.
    for (int t = tid; t < 512; t += 256) {
      int base = t * 8;
      int c = base >> 6, dh0 = base & 63;
      size_t row = (size_t)b * N_ + (size_t)j * C_ + c;
      union { uint4 u; __bf16 e[8]; } ku, vu;
      ku.u = *(const uint4*)&kvb[row * KVW_ + h * DH_ + dh0];
      vu.u = *(const uint4*)&kvb[row * KVW_ + H_ * DH_ + h * DH_ + dh0];
      *(uint4*)&s_k[base] = ku.u;
      *(uint4*)&s_vdh[base] = vu.u;
#pragma unroll
      for (int e = 0; e < 8; ++e) s_kT[(dh0 + e) * 64 + c] = ku.e[e];
    }
    if (tid < 64)
      s_lr[tid] = lrbuf[((size_t)b * N_ + (size_t)j * C_ + tid) * H_ + h];
    float momj   = mombuf[((size_t)b * NC_ + j) * H_ + h];
    float gdecay = 1.0f - decaybuf[((size_t)b * NC_ + j) * H_ + h];
    __syncthreads();

    {  // hpre = k @ w1 ; a = silu(hpre)
      v8f acc[2] = {};
      mm64(s_k, s_w1T, lane, wm, wn, acc);
#pragma unroll
      for (int tn = 0; tn < 2; ++tn)
#pragma unroll
        for (int r = 0; r < 8; ++r) {
          int mrow = wm + r + lhi, ncol = wn + tn * 16 + l15;
          float hv = acc[tn][r];
          float av = hv * sigf_(hv);
          s_hb[mrow * 64 + ncol] = (__bf16)hv;
          s_a[mrow * 64 + ncol] = (__bf16)av;
          s_aT[ncol * 64 + mrow] = (__bf16)av;
        }
    }
    __syncthreads();
    {  // pred = a @ w2 ; dpred = (2/DH)*lr_c*(pred - v)
      v8f acc[2] = {};
      mm64(s_a, s_w2T, lane, wm, wn, acc);
#pragma unroll
      for (int tn = 0; tn < 2; ++tn)
#pragma unroll
        for (int r = 0; r < 8; ++r) {
          int mrow = wm + r + lhi, ncol = wn + tn * 16 + l15;
          float d = (2.0f / (float)DH_) * s_lr[mrow] *
                    (acc[tn][r] - (float)s_vdh[mrow * 64 + ncol]);
          s_dp[mrow * 64 + ncol] = (__bf16)d;
          s_dpT[ncol * 64 + mrow] = (__bf16)d;
        }
    }
    __syncthreads();
    {  // g2 = a^T @ dpred ; m2 = mom*m2 - g2
      v8f acc[2] = {};
      mm64(s_aT, s_dpT, lane, wm, wn, acc);
#pragma unroll
      for (int tn = 0; tn < 2; ++tn)
#pragma unroll
        for (int r = 0; r < 8; ++r) {
          int p = (wm + r + lhi) * 64 + wn + tn * 16 + l15;
          s_m2[p] = momj * s_m2[p] - acc[tn][r];
        }
    }
    {  // da = dpred @ w2^T ; dh = da * silu'(hpre) -> stored transposed
      v8f acc[2] = {};
      mm64(s_dp, s_w2, lane, wm, wn, acc);  // Bt of w2^T == w2 row-major
#pragma unroll
      for (int tn = 0; tn < 2; ++tn)
#pragma unroll
        for (int r = 0; r < 8; ++r) {
          int mrow = wm + r + lhi, ncol = wn + tn * 16 + l15;
          float hv = (float)s_hb[mrow * 64 + ncol];
          float sg = sigf_(hv);
          s_vdh[ncol * 64 + mrow] =
              (__bf16)(acc[tn][r] * (sg * (1.0f + hv * (1.0f - sg))));
        }
    }
    __syncthreads();
    {  // g1 = k^T @ dh ; m1 = mom*m1 - g1
      v8f acc[2] = {};
      mm64(s_kT, s_vdh, lane, wm, wn, acc);
#pragma unroll
      for (int tn = 0; tn < 2; ++tn)
#pragma unroll
        for (int r = 0; r < 8; ++r) {
          int p = (wm + r + lhi) * 64 + wn + tn * 16 + l15;
          s_m1[p] = momj * s_m1[p] - acc[tn][r];
        }
    }
    __syncthreads();
    // W = (1-decay_j)*W + m
    for (int i = tid; i < 4096; i += 256) {
      s_W1[i] = gdecay * s_W1[i] + s_m1[i];
      s_W2[i] = gdecay * s_W2[i] + s_m2[i];
    }
    __syncthreads();
  }
}

// Retrieval per chunk: vals = silu(q@W1)@W2, multihead rmsnorm * (gamma+1) * gate
__global__ __launch_bounds__(256) void retrieve_kernel(
    const __bf16* __restrict__ qb, const __bf16* __restrict__ w1c,
    const __bf16* __restrict__ w2c, const float* __restrict__ gatebuf,
    const float* __restrict__ mh_gamma, __bf16* __restrict__ valsb) {
  __shared__ __bf16 s_q[4096], s_w1T[4096], s_w2T[4096], s_hid[4096];
  __shared__ float s_o[4096];
  __shared__ float s_rs[64];
  const int idx = blockIdx.x;  // bh*NC + j
  const int j = idx % NC_;
  const int bh = idx / NC_;
  const int b = bh / H_, h = bh % H_;
  const int tid = threadIdx.x, lane = tid & 31, wave = tid >> 5;
  const int wm = (wave >> 1) * 16, wn = (wave & 1) * 32;
  const int lhi = (lane & 16) ? 8 : 0, l15 = lane & 15;

  size_t wbase = (size_t)idx * 4096;
  // Stage: q row-major; chunk weights transposed to N-major during LDS store
  for (int t = tid; t < 512; t += 256) {
    int base = t * 8;
    int c = base >> 6, dh0 = base & 63;
    *(uint4*)&s_q[base] =
        *(const uint4*)&qb[((size_t)b * N_ + (size_t)j * C_ + c) * D_ + h * DH_ + dh0];
    union { uint4 u; __bf16 e[8]; } w1u, w2u;
    w1u.u = *(const uint4*)&w1c[wbase + base];
    w2u.u = *(const uint4*)&w2c[wbase + base];
    int k = c, n0 = dh0;  // base = k*64 + n0 in weight storage
#pragma unroll
    for (int e = 0; e < 8; ++e) {
      s_w1T[(n0 + e) * 64 + k] = w1u.e[e];
      s_w2T[(n0 + e) * 64 + k] = w2u.e[e];
    }
  }
  __syncthreads();
  {
    v8f acc[2] = {};
    mm64(s_q, s_w1T, lane, wm, wn, acc);
#pragma unroll
    for (int tn = 0; tn < 2; ++tn)
#pragma unroll
      for (int r = 0; r < 8; ++r) {
        int p = (wm + r + lhi) * 64 + wn + tn * 16 + l15;
        float hv = acc[tn][r];
        s_hid[p] = (__bf16)(hv * sigf_(hv));
      }
  }
  __syncthreads();
  {
    v8f acc[2] = {};
    mm64(s_hid, s_w2T, lane, wm, wn, acc);
#pragma unroll
    for (int tn = 0; tn < 2; ++tn)
#pragma unroll
      for (int r = 0; r < 8; ++r) {
        int p = (wm + r + lhi) * 64 + wn + tn * 16 + l15;
        s_o[p] = acc[tn][r];
      }
  }
  __syncthreads();
  if (tid < 64) {
    float ss = 0.f;
    for (int d2 = 0; d2 < 64; ++d2) { float v = s_o[tid * 64 + d2]; ss += v * v; }
    s_rs[tid] = rsqf_(ss * (1.0f / (float)DH_) + EPS_);
  }
  __syncthreads();
  for (int i = tid; i < 4096; i += 256) {
    int c = i >> 6, dh = i & 63;
    float g = gatebuf[((size_t)b * N_ + (size_t)j * C_ + c) * H_ + h];
    float gam = mh_gamma[h * DH_ + dh] + 1.0f;
    float v = s_o[i] * s_rs[c] * gam * g;
    valsb[((size_t)b * N_ + (size_t)j * C_ + c) * D_ + h * DH_ + dh] = (__bf16)v;
  }
}

extern "C" void kernel_launch(void* const* d_in, const int* in_sizes, int n_in,
                              void* d_out, int out_size, void* d_ws, size_t ws_size,
                              hipStream_t stream) {
  (void)in_sizes; (void)n_in; (void)out_size; (void)ws_size;
  const float* seq       = (const float*)d_in[0];
  const float* store_g   = (const float*)d_in[1];
  const float* retr_g    = (const float*)d_in[2];
  const float* w_kv      = (const float*)d_in[3];
  const float* w_q       = (const float*)d_in[4];
  const float* w_step    = (const float*)d_in[5];
  const float* w_mom     = (const float*)d_in[6];
  const float* w_decay   = (const float*)d_in[7];
  const float* w_gate    = (const float*)d_in[8];
  const float* w_combine = (const float*)d_in[9];
  const float* mh_gamma  = (const float*)d_in[10];
  const float* mem_w1    = (const float*)d_in[11];
  const float* mem_w2    = (const float*)d_in[12];

  size_t off = 0;
  auto alloc = [&](size_t bytes) -> void* {
    void* p = (char*)d_ws + off;
    off = (off + bytes + 255) & ~((size_t)255);
    return p;
  };
  __bf16* u_b       = (__bf16*)alloc((size_t)BN_ * D_ * 2);
  __bf16* ur_b      = (__bf16*)alloc((size_t)BN_ * D_ * 2);
  __bf16* kv_b      = (__bf16*)alloc((size_t)BN_ * KVW_ * 2);
  __bf16* q_b       = (__bf16*)alloc((size_t)BN_ * D_ * 2);
  __bf16* vals_b    = (__bf16*)alloc((size_t)BN_ * D_ * 2);
  __bf16* w1c_b     = (__bf16*)alloc((size_t)BH_ * NC_ * 4096 * 2);
  __bf16* w2c_b     = (__bf16*)alloc((size_t)BH_ * NC_ * 4096 * 2);
  __bf16* wkv_bT    = (__bf16*)alloc((size_t)D_ * KVW_ * 2);
  __bf16* wq_bT     = (__bf16*)alloc((size_t)D_ * D_ * 2);
  __bf16* wcomb_bT  = (__bf16*)alloc((size_t)D_ * D_ * 2);
  float*  chunked   = (float*)alloc((size_t)B_ * NC_ * D_ * 4);
  __bf16* chunked_b = (__bf16*)alloc((size_t)B_ * NC_ * D_ * 2);
  float*  lr_f      = (float*)alloc((size_t)BN_ * H_ * 4);
  float*  gate_f    = (float*)alloc((size_t)BN_ * H_ * 4);
  float*  decay_f   = (float*)alloc((size_t)B_ * NC_ * H_ * 4);
  float*  mom_f     = (float*)alloc((size_t)B_ * NC_ * H_ * 4);

  // 1. weight precision conversion + pre-transpose for WMMA B operand
  f2bf_t_kernel<<<(D_ * KVW_ + 255) / 256, 256, 0, stream>>>(w_kv, wkv_bT, D_, KVW_);
  f2bf_t_kernel<<<(D_ * D_ + 255) / 256, 256, 0, stream>>>(w_q, wq_bT, D_, D_);
  f2bf_t_kernel<<<(D_ * D_ + 255) / 256, 256, 0, stream>>>(w_combine, wcomb_bT, D_, D_);
  // 2. rmsnorm (store + retrieve branches)
  rmsnorm_kernel<<<BN_, 256, 0, stream>>>(seq, store_g, retr_g, u_b, ur_b);
  // 3. chunk means for decay/momentum projections
  chunk_mean_kernel<<<B_ * NC_, 256, 0, stream>>>(u_b, chunked);
  f2bf_kernel<<<(B_ * NC_ * D_ + 255) / 256, 256, 0, stream>>>(chunked, chunked_b,
                                                               B_ * NC_ * D_);
  // 4. big WMMA projections
  { dim3 g(BN_ / 128, KVW_ / 64);
    gemm_bf16_kernel<true><<<g, 256, 0, stream>>>(u_b, wkv_bT, kv_b, BN_, D_, KVW_); }
  { dim3 g(BN_ / 128, D_ / 64);
    gemm_bf16_kernel<true><<<g, 256, 0, stream>>>(ur_b, wq_bT, q_b, BN_, D_, D_); }
  // 5. small sigmoid projections (H=12 columns)
  small_proj_kernel<<<BN_ / 16, 192, 0, stream>>>(u_b, w_step, lr_f, BN_, MAX_LR_);
  small_proj_kernel<<<BN_ / 16, 192, 0, stream>>>(ur_b, w_gate, gate_f, BN_, 1.0f);
  small_proj_kernel<<<(B_ * NC_) / 16, 192, 0, stream>>>(chunked_b, w_decay, decay_f,
                                                         B_ * NC_, 1.0f);
  small_proj_kernel<<<(B_ * NC_) / 16, 192, 0, stream>>>(chunked_b, w_mom, mom_f,
                                                         B_ * NC_, 1.0f);
  // 6. fused chunk grads + momentum/decay scans
  memory_update_kernel<<<BH_, 256, 0, stream>>>(kv_b, lr_f, mom_f, decay_f, mem_w1,
                                                mem_w2, w1c_b, w2c_b);
  // 7. retrieval
  retrieve_kernel<<<BH_ * NC_, 256, 0, stream>>>(q_b, w1c_b, w2c_b, gate_f, mh_gamma,
                                                 vals_b);
  // 8. combine GEMM (f32 output)
  { dim3 g(BN_ / 128, D_ / 64);
    gemm_bf16_kernel<false><<<g, 256, 0, stream>>>(vals_b, wcomb_bT, d_out, BN_, D_, D_); }
}
